// SemanticSlotAttention_80917183857219
// MI455X (gfx1250) — matrix operand order repaired and checked by reading, not compiled, not run on hardware
//
#include <hip/hip_runtime.h>

// ---------------------------------------------------------------------------
// SemanticSlotAttention for MI455X (gfx1250, wave32, WMMA).
// All GEMMs use v_wmma_f32_16x16x32_bf16 (bf16 in, fp32 accumulate).
// bf16 values are stored as unsigned short (RNE pack) and bit-cast into
// v16bf fragments, so no __bf16 arithmetic is required.
// GEMM: 64x128 block tile, wave = 16x64 (4 accumulators, A-frag reuse 4x),
// GLOBAL_LOAD_ASYNC_TO_LDS staging (ASYNCcnt) + global_prefetch of next tile.
// ---------------------------------------------------------------------------

typedef __attribute__((ext_vector_type(16))) __bf16 v16bf;
typedef __attribute__((ext_vector_type(8)))  float  v8f;
typedef __attribute__((ext_vector_type(4)))  int    v4i;

#define Bdim   16
#define Ntok   4096
#define Kslot  64
#define Dmod   1024
#define Nhead  8
#define Dhead  128
#define ITERS  2

#if __has_builtin(__builtin_amdgcn_global_load_async_to_lds_b128) && \
    __has_builtin(__builtin_amdgcn_s_wait_asynccnt)
#define USE_ASYNC_LDS 1
#else
#define USE_ASYNC_LDS 0
#endif

__device__ __forceinline__ unsigned short f2bf(float f) {
  unsigned u = __builtin_bit_cast(unsigned, f);
  unsigned r = 0x7FFFu + ((u >> 16) & 1u);
  return (unsigned short)((u + r) >> 16);
}
__device__ __forceinline__ float bf2f(unsigned short h) {
  unsigned u = (unsigned)h << 16;
  return __builtin_bit_cast(float, u);
}

// 16-byte global->LDS tile copy: async (ASYNCcnt path) if available.
// Builtin signature (from hipcc diagnostic): (v4i AS1* src, v4i AS3* dst, imm, imm)
__device__ __forceinline__ void copy16_g2l(const unsigned short* g, unsigned short* l) {
#if USE_ASYNC_LDS
  __builtin_amdgcn_global_load_async_to_lds_b128(
      (__attribute__((address_space(1))) v4i*)(void*)g,
      (__attribute__((address_space(3))) v4i*)(void*)l, 0, 0);
#else
  *(uint4*)l = *(const uint4*)g;
#endif
}
__device__ __forceinline__ void wait_g2l() {
#if USE_ASYNC_LDS
  __builtin_amdgcn_s_wait_asynccnt(0);
#endif
}

// ---------------------------------------------------------------------------
// Generic bf16 WMMA GEMM:  C = act(alpha * A@B [+bias] [+resid])
//   block tile: 64(M) x 128(N), K-step 32; 256 threads = 8 waves,
//   wave (wm 0..3, wn 0..1) owns a 16x64 output strip -> 4 accumulators.
//   transB==1 reads logical B[k][n] = Bm[n*ldb + k]  (for X @ W.T and q @ k^T)
//   Requires M%64==0, N%128==0, Kd%32==0 (true for every call below).
// ---------------------------------------------------------------------------
__global__ __launch_bounds__(256) void gemm_bf16_wmma(
    const unsigned short* __restrict__ A, int lda, long long sA,
    const unsigned short* __restrict__ Bm, int ldb, long long sB, int transB,
    float* __restrict__ C, int ldc, long long sC,
    unsigned short* __restrict__ Cbf, int ldcbf, long long sCbf,
    const float* __restrict__ bias,
    const float* __restrict__ resid, long long sR,
    float alpha, int act, int Kd)
{
  __shared__ unsigned short As [64][40];    // [m][k], padded
  __shared__ unsigned short Bst[128][40];   // [n][k], padded (stored K-major)

  const int bz = blockIdx.z;
  const unsigned short* Ab = A + (long long)bz * sA;
  const unsigned short* Bb = Bm + (long long)bz * sB;

  const int tile_m = blockIdx.y * 64;
  const int tile_n = blockIdx.x * 128;

  const int tid  = threadIdx.x;
  const int lane = tid & 31;
  const int wid  = tid >> 5;
  const int wm   = wid >> 1;      // 0..3 : rows  [wm*16, +16)
  const int wn   = wid & 1;       // 0..1 : cols  [wn*64, +64)
  const int half = lane >> 4;     // 0/1
  const int l16  = lane & 15;

  v8f acc[4];
  acc[0] = {}; acc[1] = {}; acc[2] = {}; acc[3] = {};

  // per-thread staging coordinates
  const int a_r = (tid * 8) >> 5,  a_c = (tid * 8) & 31;       // A: 64x32
  const int bt_n = tid >> 1,       bt_c = (tid & 1) * 16;      // B^T: 128x32, 16 elems
  const int bn_r = (tid * 16) >> 7, bn_c = (tid * 16) & 127;   // B:   32x128, 16 elems

  for (int k0 = 0; k0 < Kd; k0 += 32) {
    // ---- stage A tile (64x32), 16B per thread, async if available
    const unsigned short* asrc = Ab + (long long)(tile_m + a_r) * lda + (k0 + a_c);
    copy16_g2l(asrc, &As[a_r][a_c]);

    if (transB) {   // rows of Bm are n, contiguous k: two 16B copies per thread
      const unsigned short* src = Bb + (long long)(tile_n + bt_n) * ldb + (k0 + bt_c);
      copy16_g2l(src,     &Bst[bt_n][bt_c]);
      copy16_g2l(src + 8, &Bst[bt_n][bt_c + 8]);
      if (k0 + 32 < Kd) __builtin_prefetch(src + 32, 0, 0);
    } else {        // rows of Bm are k, contiguous n; transpose into LDS
      const unsigned short* src = Bb + (long long)(k0 + bn_r) * ldb + (tile_n + bn_c);
      uint4 v0 = *(const uint4*)src;
      uint4 v1 = *(const uint4*)(src + 8);
      unsigned short tmp[16];
      *(uint4*)tmp = v0; *(uint4*)(tmp + 8) = v1;
      #pragma unroll
      for (int i = 0; i < 16; ++i) Bst[bn_c + i][bn_r] = tmp[i];
      if (k0 + 32 < Kd) __builtin_prefetch(src + 32LL * ldb, 0, 0);
    }
    if (k0 + 32 < Kd) __builtin_prefetch(asrc + 32, 0, 0);

    wait_g2l();
    __syncthreads();

    // Fragments per ISA VGPR layouts (CDNA5 ISA):
    //  A 16x32 bf16: lane<16 -> M=lane, k = i(+8 if i>=8); lane>=16 adds +8.
    //  B 32x16 bf16: n = lane&15, k = i(+8 if i>=8); lane>=16 adds +8.
    v16bf afrag;
    #pragma unroll
    for (int i = 0; i < 16; ++i) {
      int kk = (i < 8 ? i : i + 8) + half * 8;
      afrag[i] = __builtin_bit_cast(__bf16, As[wm * 16 + l16][kk]);
    }
    #pragma unroll
    for (int j = 0; j < 4; ++j) {
      v16bf bfrag;
      #pragma unroll
      for (int i = 0; i < 16; ++i) {
        int kk = (i < 8 ? i : i + 8) + half * 8;
        bfrag[i] = __builtin_bit_cast(__bf16, Bst[wn * 64 + j * 16 + l16][kk]);
      }
      acc[j] = __builtin_amdgcn_wmma_f32_16x16x32_bf16(
          false, afrag, false, bfrag, (short)0, acc[j], false, false);
    }
    __syncthreads();
  }

  // C/D f32 16x16 layout: VGPR r -> M = r + 8*half, N = lane&15.
  #pragma unroll
  for (int j = 0; j < 4; ++j) {
    #pragma unroll
    for (int r = 0; r < 8; ++r) {
      int gm = tile_m + wm * 16 + half * 8 + r;
      int gn = tile_n + wn * 64 + j * 16 + l16;
      float v = alpha * acc[j][r];
      if (bias) v += bias[gn];
      if (resid) {
        const float* rb = resid + (long long)bz * sR;
        v += rb[(long long)gm * ldc + gn];
      }
      if (act == 1)  // exact GELU
        v = 0.5f * v * (1.0f + erff(v * 0.70710678118654752f));
      if (C) {
        float* cb = C + (long long)bz * sC;
        cb[(long long)gm * ldc + gn] = v;
      }
      if (Cbf) {
        unsigned short* ob = Cbf + (long long)bz * sCbf;
        ob[(long long)gm * ldcbf + gn] = f2bf(v);
      }
    }
  }
}

// ---------------------------------------------------------------------------
// LayerNorm over D; one block per row; optional fp32 and bf16 outputs.
// ---------------------------------------------------------------------------
__global__ __launch_bounds__(256) void ln_kernel(
    const float* __restrict__ x, const float* __restrict__ g,
    const float* __restrict__ b, float* __restrict__ yf,
    unsigned short* __restrict__ ybf, int Dd)
{
  __shared__ float s1[256], s2[256];
  long long row = blockIdx.x;
  const float* xr = x + row * Dd;
  int tid = threadIdx.x;
  float sum = 0.f, sq = 0.f;
  for (int i = tid; i < Dd; i += 256) { float v = xr[i]; sum += v; sq += v * v; }
  s1[tid] = sum; s2[tid] = sq;
  __syncthreads();
  for (int off = 128; off > 0; off >>= 1) {
    if (tid < off) { s1[tid] += s1[tid + off]; s2[tid] += s2[tid + off]; }
    __syncthreads();
  }
  float mean = s1[0] / (float)Dd;
  float var  = s2[0] / (float)Dd - mean * mean;
  float inv  = rsqrtf(var + 1e-5f);
  for (int i = tid; i < Dd; i += 256) {
    float v = (xr[i] - mean) * inv * g[i] + b[i];
    if (yf)  yf[row * Dd + i]  = v;
    if (ybf) ybf[row * Dd + i] = f2bf(v);
  }
}

// ---------------------------------------------------------------------------
// Softmax over the SLOT axis (K=64) of attn[B,K,N]; writes bf16 weights.
// One thread per (b, n) column.
// ---------------------------------------------------------------------------
__global__ __launch_bounds__(256) void softmax_slots(
    const float* __restrict__ attn, unsigned short* __restrict__ aw)
{
  long long col = (long long)blockIdx.x * 256 + threadIdx.x;   // 0 .. B*N-1
  int b = (int)(col >> 12);        // N = 4096
  int n = (int)(col & 4095);
  const float* p = attn + (long long)b * Kslot * Ntok + n;
  float m = -3.4e38f;
  for (int k = 0; k < Kslot; ++k) m = fmaxf(m, p[(long long)k * Ntok]);
  float sum = 0.f;
  for (int k = 0; k < Kslot; ++k) sum += __expf(p[(long long)k * Ntok] - m);
  float inv = 1.0f / sum;
  unsigned short* o = aw + (long long)b * Kslot * Ntok + n;
  for (int k = 0; k < Kslot; ++k)
    o[(long long)k * Ntok] = f2bf(__expf(p[(long long)k * Ntok] - m) * inv);
}

// ---------------------------------------------------------------------------
// Per-(batch, head) slot self-attention: 64x64 scores, softmax over keys,
// p @ v.  qkv layout [B, K, 3D]; writes sa_bf [B, K, D] (bf16).
// ---------------------------------------------------------------------------
__global__ __launch_bounds__(256) void mha_heads(
    const float* __restrict__ qkv, unsigned short* __restrict__ sa_bf)
{
  int bh = blockIdx.x;
  int b  = bh / Nhead;
  int h  = bh % Nhead;
  __shared__ unsigned short ks[Kslot][Dhead + 4];
  __shared__ unsigned short vs[Kslot][Dhead + 4];
  __shared__ float          sc[Kslot][Kslot + 1];

  const float* base = qkv + (long long)b * Kslot * 3 * Dmod;
  for (int idx = threadIdx.x; idx < Kslot * Dhead; idx += 256) {
    int j = idx >> 7, d = idx & 127;
    ks[j][d] = f2bf(base[(long long)j * 3 * Dmod + Dmod     + h * Dhead + d]);
    vs[j][d] = f2bf(base[(long long)j * 3 * Dmod + 2 * Dmod + h * Dhead + d]);
  }
  __syncthreads();

  const float scale = 0.08838834764831845f;   // 1/sqrt(128)
  for (int idx = threadIdx.x; idx < Kslot * Kslot; idx += 256) {
    int i = idx >> 6, j = idx & 63;
    const float* q = base + (long long)i * 3 * Dmod + h * Dhead;
    float acc = 0.f;
    for (int d = 0; d < Dhead; ++d) acc += q[d] * bf2f(ks[j][d]);
    sc[i][j] = acc * scale;
  }
  __syncthreads();

  if (threadIdx.x < Kslot) {
    int i = threadIdx.x;
    float m = -3.4e38f;
    for (int j = 0; j < Kslot; ++j) m = fmaxf(m, sc[i][j]);
    float sum = 0.f;
    for (int j = 0; j < Kslot; ++j) { float e = __expf(sc[i][j] - m); sc[i][j] = e; sum += e; }
    float inv = 1.0f / sum;
    for (int j = 0; j < Kslot; ++j) sc[i][j] *= inv;
  }
  __syncthreads();

  for (int idx = threadIdx.x; idx < Kslot * Dhead; idx += 256) {
    int i = idx >> 7, d = idx & 127;
    float acc = 0.f;
    for (int j = 0; j < Kslot; ++j) acc += sc[i][j] * bf2f(vs[j][d]);
    sa_bf[(long long)b * Kslot * Dmod + (long long)i * Dmod + h * Dhead + d] = f2bf(acc);
  }
}

// ---------------------------------------------------------------------------
// Helpers: fp32 -> bf16 cast, slot-init broadcast.
// ---------------------------------------------------------------------------
__global__ __launch_bounds__(256) void cast_bf16(
    const float* __restrict__ x, unsigned short* __restrict__ y, long long n)
{
  long long i = (long long)blockIdx.x * 256 + threadIdx.x;
  if (i < n) y[i] = f2bf(x[i]);
}

__global__ __launch_bounds__(256) void bcast_slots(
    const float* __restrict__ init, float* __restrict__ S)
{
  long long i = (long long)blockIdx.x * 256 + threadIdx.x;
  S[i] = init[i & ((long long)Kslot * Dmod - 1)];   // K*D = 65536 (pow2)
}

// ---------------------------------------------------------------------------
// Launcher
// ---------------------------------------------------------------------------
static void run_gemm(hipStream_t st,
                     const unsigned short* A, int lda, long long sA,
                     const unsigned short* Bm, int ldb, long long sB, int transB,
                     float* C, int ldc, long long sC,
                     unsigned short* Cbf, int ldcbf, long long sCbf,
                     const float* bias, const float* resid, long long sR,
                     float alpha, int act, int M, int N, int Kd, int batches)
{
  dim3 grid((unsigned)(N / 128), (unsigned)(M / 64), (unsigned)batches);
  gemm_bf16_wmma<<<grid, 256, 0, st>>>(A, lda, sA, Bm, ldb, sB, transB,
                                       C, ldc, sC, Cbf, ldcbf, sCbf,
                                       bias, resid, sR, alpha, act, Kd);
}

extern "C" void kernel_launch(void* const* d_in, const int* in_sizes, int n_in,
                              void* d_out, int out_size, void* d_ws, size_t ws_size,
                              hipStream_t stream) {
  (void)in_sizes; (void)n_in; (void)out_size; (void)ws_size;
  const float* visual    = (const float*)d_in[0];
  const float* slot_init = (const float*)d_in[1];
  const float* ln_vis_g  = (const float*)d_in[2];
  const float* ln_vis_b  = (const float*)d_in[3];
  const float* ln_sl_g   = (const float*)d_in[4];
  const float* ln_sl_b   = (const float*)d_in[5];
  const float* Wq        = (const float*)d_in[6];
  const float* Wk        = (const float*)d_in[7];
  const float* Wv        = (const float*)d_in[8];
  const float* Wo        = (const float*)d_in[9];
  const float* mha_in_w  = (const float*)d_in[10];
  const float* mha_in_b  = (const float*)d_in[11];
  const float* mha_out_w = (const float*)d_in[12];
  const float* mha_out_b = (const float*)d_in[13];
  const float* ln_sa_g   = (const float*)d_in[14];
  const float* ln_sa_b   = (const float*)d_in[15];
  const float* ln_ffn_g  = (const float*)d_in[16];
  const float* ln_ffn_b  = (const float*)d_in[17];
  const float* W1        = (const float*)d_in[18];
  const float* b1        = (const float*)d_in[19];
  const float* W2        = (const float*)d_in[20];
  const float* b2        = (const float*)d_in[21];
  float* out = (float*)d_out;

  const long long DD  = (long long)Dmod * Dmod;          // 1M
  const long long BND = (long long)Bdim * Ntok * Dmod;   // 64M
  const long long BKD = (long long)Bdim * Kslot * Dmod;  // 1M
  const long long BKN = (long long)Bdim * Kslot * Ntok;  // 4M

  char* p = (char*)d_ws;
  auto carve = [&](long long bytes) {
    char* r = p; p += (bytes + 255) & ~(long long)255; return (void*)r;
  };
  unsigned short* Wq_bf   = (unsigned short*)carve(DD * 2);
  unsigned short* Wk_bf   = (unsigned short*)carve(DD * 2);
  unsigned short* Wv_bf   = (unsigned short*)carve(DD * 2);
  unsigned short* Wo_bf   = (unsigned short*)carve(DD * 2);
  unsigned short* Win_bf  = (unsigned short*)carve(3 * DD * 2);  // [3D, D], used transB
  unsigned short* Wout_bf = (unsigned short*)carve(DD * 2);      // [D, D],  used transB
  unsigned short* W1_bf   = (unsigned short*)carve(2 * DD * 2);  // [D, 2D]
  unsigned short* W2_bf   = (unsigned short*)carve(2 * DD * 2);  // [2D, D]
  unsigned short* vn_bf   = (unsigned short*)carve(BND * 2);
  unsigned short* k_bf    = (unsigned short*)carve(BND * 2);
  unsigned short* v_bf    = (unsigned short*)carve(BND * 2);
  float*          S       = (float*)carve(BKD * 4);   // slots
  float*          T       = (float*)carve(BKD * 4);   // slots after cross-attn update
  float*          U       = (float*)carve(BKD * 4);   // pre-LN (slots + mha out)
  unsigned short* sn_bf   = (unsigned short*)carve(BKD * 2);
  unsigned short* q_bf    = (unsigned short*)carve(BKD * 2);
  float*          attn    = (float*)carve(BKN * 4);
  unsigned short* aw_bf   = (unsigned short*)carve(BKN * 2);
  unsigned short* upd_bf  = (unsigned short*)carve(BKD * 2);
  unsigned short* sl_bf   = (unsigned short*)carve(BKD * 2);
  unsigned short* sa_bf   = (unsigned short*)carve(BKD * 2);
  float*          qkv     = (float*)carve((long long)Bdim * Kslot * 3 * Dmod * 4);
  unsigned short* h_bf    = (unsigned short*)carve(BKD * 2);
  unsigned short* f1_bf   = (unsigned short*)carve((long long)Bdim * Kslot * 2 * Dmod * 2);

  auto cast = [&](const float* x, unsigned short* y, long long n) {
    cast_bf16<<<(unsigned)((n + 255) / 256), 256, 0, stream>>>(x, y, n);
  };
  cast(Wq, Wq_bf, DD);       cast(Wk, Wk_bf, DD);
  cast(Wv, Wv_bf, DD);       cast(Wo, Wo_bf, DD);
  cast(mha_in_w, Win_bf, 3 * DD);
  cast(mha_out_w, Wout_bf, DD);
  cast(W1, W1_bf, 2 * DD);   cast(W2, W2_bf, 2 * DD);

  bcast_slots<<<(unsigned)(BKD / 256), 256, 0, stream>>>(slot_init, S);

  // vn = LN(visual) -> bf16
  ln_kernel<<<(unsigned)(Bdim * Ntok), 256, 0, stream>>>(
      visual, ln_vis_g, ln_vis_b, nullptr, vn_bf, Dmod);

  // k = vn @ Wk ; v = vn @ Wv   (bf16 outputs; the heavy 137-GFLOP GEMMs)
  run_gemm(stream, vn_bf, Dmod, 0, Wk_bf, Dmod, 0, 0,
           nullptr, 0, 0, k_bf, Dmod, 0, nullptr, nullptr, 0,
           1.0f, 0, Bdim * Ntok, Dmod, Dmod, 1);
  run_gemm(stream, vn_bf, Dmod, 0, Wv_bf, Dmod, 0, 0,
           nullptr, 0, 0, v_bf, Dmod, 0, nullptr, nullptr, 0,
           1.0f, 0, Bdim * Ntok, Dmod, Dmod, 1);

  const float qk_scale = 0.03125f;   // D^-0.5 = 1/32

  for (int it = 0; it < ITERS; ++it) {
    // sn = LN(slots) -> bf16 ; q = sn @ Wq -> bf16
    ln_kernel<<<(unsigned)(Bdim * Kslot), 256, 0, stream>>>(
        S, ln_sl_g, ln_sl_b, nullptr, sn_bf, Dmod);
    run_gemm(stream, sn_bf, Dmod, 0, Wq_bf, Dmod, 0, 0,
             nullptr, 0, 0, q_bf, Dmod, 0, nullptr, nullptr, 0,
             1.0f, 0, Bdim * Kslot, Dmod, Dmod, 1);

    // attn[b] = q[b] @ k[b]^T * scale   (batched, transB)
    run_gemm(stream, q_bf, Dmod, (long long)Kslot * Dmod,
             k_bf, Dmod, (long long)Ntok * Dmod, 1,
             attn, Ntok, (long long)Kslot * Ntok, nullptr, 0, 0,
             nullptr, nullptr, 0, qk_scale, 0, Kslot, Ntok, Dmod, Bdim);

    // softmax over the slot axis -> bf16 weights
    softmax_slots<<<(unsigned)(((long long)Bdim * Ntok) / 256), 256, 0, stream>>>(
        attn, aw_bf);

    // upd[b] = aw[b] @ v[b]  -> bf16
    run_gemm(stream, aw_bf, Ntok, (long long)Kslot * Ntok,
             v_bf, Dmod, (long long)Ntok * Dmod, 0,
             nullptr, 0, 0, upd_bf, Dmod, (long long)Kslot * Dmod,
             nullptr, nullptr, 0, 1.0f, 0, Kslot, Dmod, Ntok, Bdim);

    // T = S + upd @ Wo ; also bf16 copy for qkv projection
    run_gemm(stream, upd_bf, Dmod, 0, Wo_bf, Dmod, 0, 0,
             T, Dmod, 0, sl_bf, Dmod, 0, nullptr, S, 0,
             1.0f, 0, Bdim * Kslot, Dmod, Dmod, 1);

    // qkv = T @ mha_in_w^T + mha_in_b
    run_gemm(stream, sl_bf, Dmod, 0, Win_bf, Dmod, 0, 1,
             qkv, 3 * Dmod, 0, nullptr, 0, 0, mha_in_b, nullptr, 0,
             1.0f, 0, Bdim * Kslot, 3 * Dmod, Dmod, 1);

    // per-head self-attention over the 64 slots
    mha_heads<<<(unsigned)(Bdim * Nhead), 256, 0, stream>>>(qkv, sa_bf);

    // U = T + sa @ mha_out_w^T + mha_out_b ; slots = LN(U)
    run_gemm(stream, sa_bf, Dmod, 0, Wout_bf, Dmod, 0, 1,
             U, Dmod, 0, nullptr, 0, 0, mha_out_b, T, 0,
             1.0f, 0, Bdim * Kslot, Dmod, Dmod, 1);
    ln_kernel<<<(unsigned)(Bdim * Kslot), 256, 0, stream>>>(
        U, ln_sa_g, ln_sa_b, S, nullptr, Dmod);
  }

  // h = LN(slots) -> bf16 ; f1 = gelu(h @ W1 + b1) -> bf16
  ln_kernel<<<(unsigned)(Bdim * Kslot), 256, 0, stream>>>(
      S, ln_ffn_g, ln_ffn_b, nullptr, h_bf, Dmod);
  run_gemm(stream, h_bf, Dmod, 0, W1_bf, 2 * Dmod, 0, 0,
           nullptr, 0, 0, f1_bf, 2 * Dmod, 0, b1, nullptr, 0,
           1.0f, 1, Bdim * Kslot, 2 * Dmod, Dmod, 1);

  // out = slots + f1 @ W2 + b2
  run_gemm(stream, f1_bf, 2 * Dmod, 0, W2_bf, Dmod, 0, 0,
           out, Dmod, 0, nullptr, 0, 0, b2, S, 0,
           1.0f, 0, Bdim * Kslot, Dmod, 2 * Dmod, 1);
}
